// GLUBlock_53446573032203
// MI455X (gfx1250) — matrix operand order repaired and checked by reading, not compiled, not run on hardware
//
#include <hip/hip_runtime.h>

typedef __attribute__((ext_vector_type(16))) __bf16 v16bf;
typedef __attribute__((ext_vector_type(8)))  float  v8f;
typedef __attribute__((ext_vector_type(4)))  __bf16 v4bf;
typedef __attribute__((ext_vector_type(4)))  int    v4i_;

#define DIM    1024
#define NOUT   2048
#define NTOK   32768      // 8 * 4096 tokens
#define BM     128        // block tile M
#define BN     64         // block tile N (per half; V and G both computed)
#define BK     32         // K per LDS stage (== WMMA K)
#define NSTAGE (DIM / BK) // 32

#if defined(__has_builtin)
#if __has_builtin(__builtin_amdgcn_global_load_async_to_lds_b128)
#define HAVE_ASYNC_LDS 1
#endif
#endif

union FragU { uint4 q[2]; v16bf v; };

// ---------------------------------------------------------------------------
// Kernel 1: LayerNorm (fp32 stats) -> bf16 activations h
// ---------------------------------------------------------------------------
__global__ __launch_bounds__(256) void ln_bf16_kernel(
    const float* __restrict__ x, const float* __restrict__ gamma,
    const float* __restrict__ beta, __bf16* __restrict__ h)
{
    const int row  = blockIdx.x;
    const int t    = threadIdx.x;
    const int w    = t >> 5;
    const int lane = t & 31;

    const float4 v = ((const float4*)(x + (size_t)row * DIM))[t];
    float s = v.x + v.y + v.z + v.w;
    float q = v.x * v.x + v.y * v.y + v.z * v.z + v.w * v.w;
    for (int o = 16; o > 0; o >>= 1) {
        s += __shfl_xor(s, o, 32);
        q += __shfl_xor(q, o, 32);
    }
    __shared__ float ss[8], qq[8];
    if (lane == 0) { ss[w] = s; qq[w] = q; }
    __syncthreads();
    float S = 0.f, Q = 0.f;
    for (int i = 0; i < 8; ++i) { S += ss[i]; Q += qq[i]; }
    const float mean = S * (1.0f / DIM);
    const float var  = Q * (1.0f / DIM) - mean * mean;
    const float rstd = rsqrtf(var + 1e-5f);

    const float4 g4 = ((const float4*)gamma)[t];
    const float4 b4 = ((const float4*)beta)[t];
    v4bf hv;
    hv.x = (__bf16)((v.x - mean) * rstd * g4.x + b4.x);
    hv.y = (__bf16)((v.y - mean) * rstd * g4.y + b4.y);
    hv.z = (__bf16)((v.z - mean) * rstd * g4.z + b4.z);
    hv.w = (__bf16)((v.w - mean) * rstd * g4.w + b4.w);
    ((v4bf*)(h + (size_t)row * DIM))[t] = hv;
}

// ---------------------------------------------------------------------------
// Kernel 2: W [K,N] fp32 -> Wt [N,K] bf16 (transpose+convert), tiled via LDS
// ---------------------------------------------------------------------------
__global__ __launch_bounds__(256) void wt_kernel(
    const float* __restrict__ W, __bf16* __restrict__ Wt)
{
    __shared__ float tile[32][33];
    const int k0 = blockIdx.x * 32;
    const int n0 = blockIdx.y * 32;
    const int tx = threadIdx.x & 31;
    const int ty = threadIdx.x >> 5;   // 32x8 logical layout
    for (int j = 0; j < 4; ++j)
        tile[ty + 8 * j][tx] = W[(size_t)(k0 + ty + 8 * j) * NOUT + n0 + tx];
    __syncthreads();
    for (int j = 0; j < 4; ++j)
        Wt[(size_t)(n0 + ty + 8 * j) * DIM + k0 + tx] = (__bf16)tile[tx][ty + 8 * j];
}

// ---------------------------------------------------------------------------
// Staging helpers: async global->LDS DMA (ASYNCcnt) when available,
// otherwise reg-staged global->VGPR->ds_store path.
// ---------------------------------------------------------------------------
struct StageRegs { uint4 a0, a1, b0, b1; };

#if defined(HAVE_ASYNC_LDS)
__device__ __forceinline__ void async_b128(const void* g, void* l)
{
    __builtin_amdgcn_global_load_async_to_lds_b128(
        (__attribute__((address_space(1))) v4i_*)g,
        (__attribute__((address_space(3))) v4i_*)l, 0, 0);
}
#endif

__device__ __forceinline__ void stage_issue(
    StageRegs& r,
    const __bf16* aS0, const __bf16* aS1,
    const __bf16* bS0, const __bf16* bS1,
    __bf16* lA0, __bf16* lA1, __bf16* lB0, __bf16* lB1)
{
#if defined(HAVE_ASYNC_LDS)
    async_b128(aS0, lA0);
    async_b128(aS1, lA1);
    async_b128(bS0, lB0);
    async_b128(bS1, lB1);
    (void)r;
#else
    r.a0 = *(const uint4*)aS0;
    r.a1 = *(const uint4*)aS1;
    r.b0 = *(const uint4*)bS0;
    r.b1 = *(const uint4*)bS1;
    (void)lA0; (void)lA1; (void)lB0; (void)lB1;
#endif
}

__device__ __forceinline__ void stage_commit(
    StageRegs& r,
    __bf16* lA0, __bf16* lA1, __bf16* lB0, __bf16* lB1)
{
#if defined(HAVE_ASYNC_LDS)
#if defined(__has_builtin) && __has_builtin(__builtin_amdgcn_s_wait_asynccnt)
    __builtin_amdgcn_s_wait_asynccnt(0);
#else
    asm volatile("s_wait_asynccnt 0x0" ::: "memory");
#endif
    (void)r; (void)lA0; (void)lA1; (void)lB0; (void)lB1;
#else
    *(uint4*)lA0 = r.a0;
    *(uint4*)lA1 = r.a1;
    *(uint4*)lB0 = r.b0;
    *(uint4*)lB1 = r.b1;
#endif
}

// ---------------------------------------------------------------------------
// Kernel 3: fused bf16 WMMA GEMM + GLU epilogue (value & gate halves together)
// ---------------------------------------------------------------------------
__global__ __launch_bounds__(256) void glu_gemm_kernel(
    const __bf16* __restrict__ h, const __bf16* __restrict__ Wt,
    const float* __restrict__ x, const float* __restrict__ bias,
    float* __restrict__ out)
{
    __shared__ __align__(16) __bf16 lsA[2][BM * BK];       // 16 KB
    __shared__ __align__(16) __bf16 lsB[2][2 * BN * BK];   // 16 KB (V rows 0-63, G rows 64-127)

    const int t    = threadIdx.x;
    const int lane = t & 31;
    const int w    = t >> 5;
    const int wm   = w >> 1;   // 0..3 -> 32-row strip
    const int wn   = w & 1;    // 0..1 -> 32-col strip
    const int gM0  = blockIdx.y * BM;
    const int gN0  = blockIdx.x * BN;

    // --- global staging: 512 uint4 per tile per stage; 2 uint4/thread each ---
    const int ia0 = t, ia1 = t + 256;
    const int rowA0 = ia0 >> 2, rowA1 = ia1 >> 2;          // 0..127
    const int koA0  = (ia0 & 3) << 3, koA1 = (ia1 & 3) << 3;
    const __bf16* aSrc0 = h + (size_t)(gM0 + rowA0) * DIM + koA0;
    const __bf16* aSrc1 = h + (size_t)(gM0 + rowA1) * DIM + koA1;
    const int nG0 = (rowA0 < BN) ? (gN0 + rowA0) : (DIM + gN0 + rowA0 - BN);
    const int nG1 = (rowA1 < BN) ? (gN0 + rowA1) : (DIM + gN0 + rowA1 - BN);
    const __bf16* bSrc0 = Wt + (size_t)nG0 * DIM + koA0;
    const __bf16* bSrc1 = Wt + (size_t)nG1 * DIM + koA1;

    // --- fragment LDS addressing (uint4 indices) ---
    const int r16 = lane & 15;
    const int hg  = lane >> 4;
    int aIdx[2][2], bIdxV[2][2], bIdxG[2][2];
    for (int mt = 0; mt < 2; ++mt) {
        // A 16-bit layout: lanes<16 K {0-7,16-23}; lanes>=16 K {8-15,24-31}
        const int base = (wm * 32 + mt * 16 + r16) * BK + hg * 8;
        aIdx[mt][0] = base >> 3;
        aIdx[mt][1] = (base + 16) >> 3;
    }
    for (int nt = 0; nt < 2; ++nt) {
        // B 16-bit layout: lanes<16 K {0-7,16-23}; lanes>=16 K {8-15,24-31}
        const int baseV = (wn * 32 + nt * 16 + r16) * BK + hg * 8;
        const int baseG = (BN + wn * 32 + nt * 16 + r16) * BK + hg * 8;
        bIdxV[nt][0] = baseV >> 3; bIdxV[nt][1] = (baseV >> 3) + 2;
        bIdxG[nt][0] = baseG >> 3; bIdxG[nt][1] = (baseG >> 3) + 2;
    }

    v8f accV[2][2] = {};
    v8f accG[2][2] = {};

    // prologue: fill stage 0
    {
        StageRegs r;
        stage_issue(r, aSrc0, aSrc1, bSrc0, bSrc1,
                    &lsA[0][ia0 * 8], &lsA[0][ia1 * 8],
                    &lsB[0][ia0 * 8], &lsB[0][ia1 * 8]);
        stage_commit(r, &lsA[0][ia0 * 8], &lsA[0][ia1 * 8],
                     &lsB[0][ia0 * 8], &lsB[0][ia1 * 8]);
    }
    __syncthreads();

    for (int kk = 0; kk < NSTAGE; ++kk) {
        const int cur = kk & 1;
        const int nb  = cur ^ 1;
        StageRegs r;
        if (kk + 1 < NSTAGE) {
            const int k0 = (kk + 1) * BK;
            stage_issue(r, aSrc0 + k0, aSrc1 + k0, bSrc0 + k0, bSrc1 + k0,
                        &lsA[nb][ia0 * 8], &lsA[nb][ia1 * 8],
                        &lsB[nb][ia0 * 8], &lsB[nb][ia1 * 8]);
        }

        const uint4* A = (const uint4*)lsA[cur];
        const uint4* B = (const uint4*)lsB[cur];
        v16bf af[2], bV[2], bG[2];
        for (int mt = 0; mt < 2; ++mt) {
            FragU u; u.q[0] = A[aIdx[mt][0]]; u.q[1] = A[aIdx[mt][1]];
            af[mt] = u.v;
        }
        for (int nt = 0; nt < 2; ++nt) {
            FragU u;  u.q[0]  = B[bIdxV[nt][0]]; u.q[1]  = B[bIdxV[nt][1]];
            bV[nt] = u.v;
            FragU u2; u2.q[0] = B[bIdxG[nt][0]]; u2.q[1] = B[bIdxG[nt][1]];
            bG[nt] = u2.v;
        }
        for (int mt = 0; mt < 2; ++mt)
            for (int nt = 0; nt < 2; ++nt) {
                accV[mt][nt] = __builtin_amdgcn_wmma_f32_16x16x32_bf16(
                    false, af[mt], false, bV[nt], (short)0, accV[mt][nt], false, false);
                accG[mt][nt] = __builtin_amdgcn_wmma_f32_16x16x32_bf16(
                    false, af[mt], false, bG[nt], (short)0, accG[mt][nt], false, false);
            }

        if (kk + 1 < NSTAGE) {
            stage_commit(r, &lsA[nb][ia0 * 8], &lsA[nb][ia1 * 8],
                         &lsB[nb][ia0 * 8], &lsB[nb][ia1 * 8]);
        }
        __syncthreads();
    }

    // --- epilogue: D layout: lane -> N = lane%16; VGPR j -> M = j + 8*(lane/16)
    const int ncol0 = gN0 + wn * 32 + r16;
    const int mrow0 = gM0 + wm * 32 + (hg << 3);
    for (int nt = 0; nt < 2; ++nt) {
        const int col = ncol0 + nt * 16;
        const float bv = bias[col];
        const float bg = bias[DIM + col];
        for (int mt = 0; mt < 2; ++mt) {
            const int rbase = mrow0 + mt * 16;
            const v8f vv = accV[mt][nt];
            const v8f gg = accG[mt][nt];
            for (int j = 0; j < 8; ++j) {
                const size_t idx = (size_t)(rbase + j) * DIM + col;
                const float val  = vv[j] + bv;
                const float gate = gg[j] + bg;
                const float sg   = 1.0f / (1.0f + __expf(-gate));
                out[idx] = x[idx] + val * sg;
            }
        }
    }
}

extern "C" void kernel_launch(void* const* d_in, const int* in_sizes, int n_in,
                              void* d_out, int out_size, void* d_ws, size_t ws_size,
                              hipStream_t stream) {
    const float* x     = (const float*)d_in[0];
    const float* gamma = (const float*)d_in[1];
    const float* beta  = (const float*)d_in[2];
    const float* W     = (const float*)d_in[3];
    const float* bias  = (const float*)d_in[4];
    float* out = (float*)d_out;

    // workspace: h bf16 [32768,1024] (64 MB) then Wt bf16 [2048,1024] (4 MB)
    __bf16* h  = (__bf16*)d_ws;
    __bf16* Wt = (__bf16*)((char*)d_ws + (size_t)NTOK * DIM * sizeof(__bf16));

    ln_bf16_kernel<<<NTOK, 256, 0, stream>>>(x, gamma, beta, h);
    wt_kernel<<<dim3(DIM / 32, NOUT / 32), 256, 0, stream>>>(W, Wt);
    glu_gemm_kernel<<<dim3(DIM / BN, NTOK / BM), 256, 0, stream>>>(h, Wt, x, bias, out);
}
